// Pose3D_58110907515333
// MI455X (gfx1250) — matrix operand order repaired and checked by reading, compiled, not run on hardware
//
#include <hip/hip_runtime.h>
#include <hip/hip_bf16.h>
#include <math.h>

// ---- problem constants ----
#define BB    32
#define FF    243
#define PT    17        // tokens
#define PP    32        // padded tokens (2 WMMA row tiles)
#define CINN  2
#define CCH   128       // channels
#define NHD   8
#define HDD   16
#define HIDD  256
#define NLAY  4
#define NSEQ  (BB*FF)   // 7776
#define LNEPS 1e-6f

typedef __bf16 bf16_t;
typedef bf16_t bf16x16 __attribute__((ext_vector_type(16)));
typedef float  f32x8   __attribute__((ext_vector_type(8)));

// ---- bf16 weight-fragment workspace layout (element offsets) ----
#define TILE_ELEMS 512                    // one 32x16 B tile, fragment order
#define QKV_TPL  (4*24)                   // ktiles * ntiles per layer
#define PROJ_TPL (4*8)
#define FC1_TPL  (4*16)
#define FC2_TPL  (8*8)
#define WS_QKV  0
#define WS_PROJ (WS_QKV  + NLAY*QKV_TPL *TILE_ELEMS)
#define WS_FC1  (WS_PROJ + NLAY*PROJ_TPL*TILE_ELEMS)
#define WS_FC2  (WS_FC1  + NLAY*FC1_TPL *TILE_ELEMS)

// ---------------------------------------------------------------------------
// Prep: swizzle fp32 row-major (Krows, Ncols) weights into bf16 WMMA B-fragment
// tiles. One wave per 32x16 tile; each lane writes its contiguous 16 elements.
// B k order used here (contiguous, same as load_afrag): elem e of lane: K = kt*32 + (lane>=16?16:0) + e,
// N = nt*16 + (lane&15).
// ---------------------------------------------------------------------------
__global__ void prep_wtiles(const float* __restrict__ src, bf16_t* __restrict__ dst,
                            int NT, int Ncols) {
    int t    = blockIdx.x;
    int lane = threadIdx.x;
    int kt = t / NT, nt = t % NT;
    int n  = nt * 16 + (lane & 15);
    int kb = kt * 32 + (lane & 16);     // +16 for upper half-wave
    bf16x16 v;
#pragma unroll
    for (int e = 0; e < 16; ++e)
        v[e] = (bf16_t)src[(size_t)(kb + e) * Ncols + n];
    *(bf16x16*)(dst + (size_t)t * TILE_ELEMS + lane * 16) = v;
}

// ---------------------------------------------------------------------------
// A-fragment from fp32 LDS row-major [rows][lda]:
// lanes 0-15 row M=lane: K=kb+e; lanes 16-31: kb+=16
// kb is a multiple of 16 -> float4 (ds_load_b128) friendly.
// ---------------------------------------------------------------------------
__device__ __forceinline__ bf16x16 load_afrag(const float* A, int lda,
                                              int mt, int kt, int lane) {
    int row = mt * 16 + (lane & 15);
    int kb  = kt * 32 + (lane & 16);
    const float* p = A + row * lda + kb;
    float4 a0 = *(const float4*)(p + 0);
    float4 a1 = *(const float4*)(p + 4);
    float4 a2 = *(const float4*)(p + 8);
    float4 a3 = *(const float4*)(p + 12);
    bf16x16 v;
    v[0]  = (bf16_t)a0.x; v[1]  = (bf16_t)a0.y; v[2]  = (bf16_t)a0.z; v[3]  = (bf16_t)a0.w;
    v[4]  = (bf16_t)a1.x; v[5]  = (bf16_t)a1.y; v[6]  = (bf16_t)a1.z; v[7]  = (bf16_t)a1.w;
    v[8]  = (bf16_t)a2.x; v[9]  = (bf16_t)a2.y; v[10] = (bf16_t)a2.z; v[11] = (bf16_t)a2.w;
    v[12] = (bf16_t)a3.x; v[13] = (bf16_t)a3.y; v[14] = (bf16_t)a3.z; v[15] = (bf16_t)a3.w;
    return v;
}

// MODE: 0 = bias+store, 1 = bias+accumulate-into-dest, 2 = bias+GELU+store
// Each wave owns column strip nt = wave, wave+8, ... and computes BOTH mt row
// tiles per loaded B fragment (two accumulators) -> every weight tile is
// fetched exactly once per block. For KT<=4 both mt's A fragments are cached
// in registers; for KT==8 they are reloaded per k-step (LDS is cheap, L2 not).
template <int KT, int NT, int MODE>
__device__ __forceinline__ void gemm_wmma(const float* __restrict__ A, int lda,
                                          const bf16_t* __restrict__ W,
                                          const float* __restrict__ bias,
                                          float* __restrict__ C, int ldc, int tid) {
    int lane = tid & 31, wave = tid >> 5;
    constexpr bool CACHEA = (KT <= 4);
    bf16x16 afr[CACHEA ? 2 * KT : 1];
    if constexpr (CACHEA) {
#pragma unroll
        for (int kt = 0; kt < KT; ++kt) {
            afr[kt]      = load_afrag(A, lda, 0, kt, lane);
            afr[KT + kt] = load_afrag(A, lda, 1, kt, lane);
        }
    }
    for (int nt = wave; nt < NT; nt += 8) {
        f32x8 acc0 = {}, acc1 = {};
#pragma unroll
        for (int kt = 0; kt < KT; ++kt) {
            bf16x16 b = *(const bf16x16*)(W + ((size_t)(kt * NT + nt) * 32 + lane) * 16);
            bf16x16 a0, a1;
            if constexpr (CACHEA) { a0 = afr[kt]; a1 = afr[KT + kt]; }
            else { a0 = load_afrag(A, lda, 0, kt, lane); a1 = load_afrag(A, lda, 1, kt, lane); }
            acc0 = __builtin_amdgcn_wmma_f32_16x16x32_bf16(
                false, a0, false, b, (short)0, acc0, false, false);
            acc1 = __builtin_amdgcn_wmma_f32_16x16x32_bf16(
                false, a1, false, b, (short)0, acc1, false, false);
        }
        // D layout: VGPR r -> lanes 0-15: M=r, lanes 16-31: M=8+r; N=lane&15
        int row = (lane >> 4) << 3;
        int col = nt * 16 + (lane & 15);
        float bv = bias[col];
#pragma unroll
        for (int r = 0; r < 8; ++r) {
            float v = acc0[r] + bv;
            if (MODE == 2) v = 0.5f * v * (1.0f + erff(v * 0.70710678118654752f));
            if (MODE == 1) v += C[(row + r) * ldc + col];
            C[(row + r) * ldc + col] = v;
        }
#pragma unroll
        for (int r = 0; r < 8; ++r) {
            float v = acc1[r] + bv;
            if (MODE == 2) v = 0.5f * v * (1.0f + erff(v * 0.70710678118654752f));
            if (MODE == 1) v += C[(16 + row + r) * ldc + col];
            C[(16 + row + r) * ldc + col] = v;
        }
    }
}

__device__ __forceinline__ void layernorm_rows(const float* __restrict__ src,
                                               float* __restrict__ dst,
                                               const float* __restrict__ g,
                                               const float* __restrict__ b,
                                               int lane, int wave) {
    for (int p = wave; p < PT; p += 8) {
        const float* r = src + p * CCH;
        float v[4], s0 = 0.f, s1 = 0.f;
#pragma unroll
        for (int e = 0; e < 4; ++e) { v[e] = r[lane * 4 + e]; s0 += v[e]; s1 += v[e] * v[e]; }
#pragma unroll
        for (int off = 16; off; off >>= 1) {
            s0 += __shfl_xor(s0, off, 32);
            s1 += __shfl_xor(s1, off, 32);
        }
        float m  = s0 * (1.0f / CCH);
        float rs = rsqrtf(s1 * (1.0f / CCH) - m * m + LNEPS);
#pragma unroll
        for (int e = 0; e < 4; ++e) {
            int c = lane * 4 + e;
            dst[p * CCH + c] = (v[e] - m) * rs * g[c] + b[c];
        }
    }
}

// ---------------------------------------------------------------------------
// Fused 4-layer gated transformer, one block (8 wave32) per sequence.
// ---------------------------------------------------------------------------
__global__ __launch_bounds__(256)
void pose3d_fused(const float* __restrict__ x,   const float* __restrict__ adj,
                  const float* __restrict__ w_embed, const float* __restrict__ b_embed,
                  const float* __restrict__ pos,
                  const float* __restrict__ ln1_g, const float* __restrict__ ln1_b,
                  const float* __restrict__ b_qkv, const float* __restrict__ w_gate,
                  const float* __restrict__ b_gate,
                  const float* __restrict__ b_proj,
                  const float* __restrict__ ln2_g, const float* __restrict__ ln2_b,
                  const float* __restrict__ b_fc1, const float* __restrict__ b_fc2,
                  const float* __restrict__ fn_g,  const float* __restrict__ fn_b,
                  const bf16_t* __restrict__ wsw,  float* __restrict__ out) {
    __shared__ float sh_h[PP * CCH];       // residual stream  (16 KB)
    __shared__ float sh_y[PP * CCH];       // LN out / attn out (16 KB)
    __shared__ float sh_big[PP * 3 * CCH]; // qkv, reused as MLP hidden (48 KB)
    __shared__ float sh_gate[PT];
    __shared__ float sh_adj[PT * PT];

    int tid  = threadIdx.x;
    int lane = tid & 31, wave = tid >> 5;
    int s    = blockIdx.x;
    const float* xs = x + (size_t)s * PT * CINN;

    for (int i = tid; i < PT * PT; i += 256) sh_adj[i] = adj[i];
    // embed: h = x @ w_embed + b_embed + pos   (K=2, scalar)
    for (int i = tid; i < PT * CCH; i += 256) {
        int p = i / CCH, c = i - p * CCH;
        sh_h[i] = xs[p * CINN + 0] * w_embed[c] + xs[p * CINN + 1] * w_embed[CCH + c]
                + b_embed[c] + pos[i];
    }
    __syncthreads();

    const bf16_t* wqkv  = wsw + WS_QKV;
    const bf16_t* wproj = wsw + WS_PROJ;
    const bf16_t* wfc1  = wsw + WS_FC1;
    const bf16_t* wfc2  = wsw + WS_FC2;
    const float scale = 0.25f;             // hd^-0.5, hd=16

    for (int L = 0; L < NLAY; ++L) {
        // ---- LN1 -> y ----
        layernorm_rows(sh_h, sh_y, ln1_g + L * CCH, ln1_b + L * CCH, lane, wave);
        __syncthreads();

        // ---- qkv = y @ Wqkv + b   (17x128 . 128x384) ----
        gemm_wmma<4, 24, 0>(sh_y, CCH, wqkv + (size_t)L * QKV_TPL * TILE_ELEMS,
                            b_qkv + L * 3 * CCH, sh_big, 3 * CCH, tid);

        // ---- gate[p] = sigmoid(y[p] . w_gate + b) (disjoint from qkv writes) ----
        {
            const float* wg = w_gate + L * CCH;
            for (int p = wave; p < PT; p += 8) {
                const float* r = sh_y + p * CCH;
                float d = 0.f;
#pragma unroll
                for (int e = 0; e < 4; ++e) d += r[lane * 4 + e] * wg[lane * 4 + e];
#pragma unroll
                for (int off = 16; off; off >>= 1) d += __shfl_xor(d, off, 32);
                if (lane == 0) sh_gate[p] = 1.0f / (1.0f + __expf(-(d + b_gate[L])));
            }
        }
        __syncthreads();

        // ---- gated attention, one head per wave; o -> sh_y ----
        {
            int hh = wave;
            for (int n = 0; n < PT; ++n) {
                int m = lane;
                float sval;
                if (m < PT) {
                    float d = 0.f;
#pragma unroll
                    for (int dd = 0; dd < HDD; ++dd)
                        d += sh_big[n * 384 + hh * HDD + dd] *
                             sh_big[m * 384 + CCH + hh * HDD + dd];
                    float g = sh_gate[m];                     // gate indexed by KEY
                    sval = d * scale * (1.0f - g) + sh_adj[n * PT + m] * g;
                } else sval = -3.0e38f;
                float mx = sval;
#pragma unroll
                for (int off = 16; off; off >>= 1) mx = fmaxf(mx, __shfl_xor(mx, off, 32));
                float pe = (m < PT) ? __expf(sval - mx) : 0.f;
                float sm = pe;
#pragma unroll
                for (int off = 16; off; off >>= 1) sm += __shfl_xor(sm, off, 32);
                float prob = pe / sm;
                float o = 0.f;
                for (int mm = 0; mm < PT; ++mm) {
                    float pm = __shfl(prob, mm, 32);
                    o += pm * sh_big[mm * 384 + 2 * CCH + hh * HDD + (lane & 15)];
                }
                if (lane < HDD) sh_y[n * CCH + hh * HDD + lane] = o;
            }
        }
        __syncthreads();

        // ---- h += o @ Wproj + b ----
        gemm_wmma<4, 8, 1>(sh_y, CCH, wproj + (size_t)L * PROJ_TPL * TILE_ELEMS,
                           b_proj + L * CCH, sh_h, CCH, tid);
        __syncthreads();

        // ---- LN2 -> y ----
        layernorm_rows(sh_h, sh_y, ln2_g + L * CCH, ln2_b + L * CCH, lane, wave);
        __syncthreads();

        // ---- hid = gelu(y @ Wfc1 + b) ----
        gemm_wmma<4, 16, 2>(sh_y, CCH, wfc1 + (size_t)L * FC1_TPL * TILE_ELEMS,
                            b_fc1 + L * HIDD, sh_big, HIDD, tid);
        __syncthreads();

        // ---- h += hid @ Wfc2 + b ----
        gemm_wmma<8, 8, 1>(sh_big, HIDD, wfc2 + (size_t)L * FC2_TPL * TILE_ELEMS,
                           b_fc2 + L * CCH, sh_h, CCH, tid);
        __syncthreads();
    }

    // ---- final LN straight to global ----
    for (int p = wave; p < PT; p += 8) {
        const float* r = sh_h + p * CCH;
        float v[4], s0 = 0.f, s1 = 0.f;
#pragma unroll
        for (int e = 0; e < 4; ++e) { v[e] = r[lane * 4 + e]; s0 += v[e]; s1 += v[e] * v[e]; }
#pragma unroll
        for (int off = 16; off; off >>= 1) {
            s0 += __shfl_xor(s0, off, 32);
            s1 += __shfl_xor(s1, off, 32);
        }
        float m  = s0 * (1.0f / CCH);
        float rs = rsqrtf(s1 * (1.0f / CCH) - m * m + LNEPS);
        float* op = out + ((size_t)s * PT + p) * CCH;
#pragma unroll
        for (int e = 0; e < 4; ++e) {
            int c = lane * 4 + e;
            op[c] = (v[e] - m) * rs * fn_g[c] + fn_b[c];
        }
    }
}

extern "C" void kernel_launch(void* const* d_in, const int* in_sizes, int n_in,
                              void* d_out, int out_size, void* d_ws, size_t ws_size,
                              hipStream_t stream) {
    const float* x       = (const float*)d_in[0];
    const float* adj     = (const float*)d_in[1];
    const float* w_embed = (const float*)d_in[2];
    const float* b_embed = (const float*)d_in[3];
    const float* pos     = (const float*)d_in[4];
    const float* ln1_g   = (const float*)d_in[5];
    const float* ln1_b   = (const float*)d_in[6];
    const float* w_qkv   = (const float*)d_in[7];
    const float* b_qkv   = (const float*)d_in[8];
    const float* w_gate  = (const float*)d_in[9];
    const float* b_gate  = (const float*)d_in[10];
    const float* w_proj  = (const float*)d_in[11];
    const float* b_proj  = (const float*)d_in[12];
    const float* ln2_g   = (const float*)d_in[13];
    const float* ln2_b   = (const float*)d_in[14];
    const float* w_fc1   = (const float*)d_in[15];
    const float* b_fc1   = (const float*)d_in[16];
    const float* w_fc2   = (const float*)d_in[17];
    const float* b_fc2   = (const float*)d_in[18];
    const float* fn_g    = (const float*)d_in[19];
    const float* fn_b    = (const float*)d_in[20];
    float* out  = (float*)d_out;
    bf16_t* wsb = (bf16_t*)d_ws;

    // swizzle fp32 weights into bf16 WMMA B-fragment tiles (L2-resident, ~1 MB)
    prep_wtiles<<<NLAY * QKV_TPL,  32, 0, stream>>>(w_qkv,  wsb + WS_QKV,  24, 3 * CCH);
    prep_wtiles<<<NLAY * PROJ_TPL, 32, 0, stream>>>(w_proj, wsb + WS_PROJ, 8,  CCH);
    prep_wtiles<<<NLAY * FC1_TPL,  32, 0, stream>>>(w_fc1,  wsb + WS_FC1,  16, HIDD);
    prep_wtiles<<<NLAY * FC2_TPL,  32, 0, stream>>>(w_fc2,  wsb + WS_FC2,  8,  CCH);

    pose3d_fused<<<NSEQ, 256, 0, stream>>>(
        x, adj, w_embed, b_embed, pos, ln1_g, ln1_b, b_qkv, w_gate, b_gate,
        b_proj, ln2_g, ln2_b, b_fc1, b_fc2, fn_g, fn_b, wsb, out);
}
